// TransGeneratorFeatureList_47373489275302
// MI455X (gfx1250) — matrix operand order repaired and checked by reading, not compiled, not run on hardware
//
#include <hip/hip_runtime.h>

// Problem constants
#define EE 1024
#define HH 16
#define FF 4096
#define SS 1024
#define BB 4
#define LL 6
#define DH 64
#define PADTOK 1023

typedef __attribute__((ext_vector_type(16))) __bf16 v16bf;
typedef __attribute__((ext_vector_type(8)))  float  v8f;
typedef __attribute__((ext_vector_type(4)))  int    v4i;

#ifndef __has_builtin
#define __has_builtin(x) 0
#endif
#if __has_builtin(__builtin_amdgcn_global_load_async_to_lds_b128) && \
    __has_builtin(__builtin_amdgcn_s_wait_asynccnt)
#define HAVE_ASYNC_LDS 1
#else
#define HAVE_ASYNC_LDS 0
#endif

static __device__ __forceinline__ __bf16 f2bf(float f) {
  unsigned u = __builtin_bit_cast(unsigned, f);
  u += 0x7FFFu + ((u >> 16) & 1u);          // round-to-nearest-even
  unsigned short h = (unsigned short)(u >> 16);
  return __builtin_bit_cast(__bf16, h);
}

static __device__ __forceinline__ v8f vzero8() {
  v8f z;
#pragma unroll
  for (int i = 0; i < 8; ++i) z[i] = 0.0f;
  return z;
}

static __device__ __forceinline__ float gelu_exact(float x) {
  return 0.5f * x * (1.0f + erff(x * 0.70710678118654752f));
}

// 16-byte async copy global -> LDS (CDNA5 async path, ASYNCcnt-tracked)
#if HAVE_ASYNC_LDS
static __device__ __forceinline__ void cp16_async(const __bf16* g, __bf16* l) {
  __builtin_amdgcn_global_load_async_to_lds_b128(
      (__attribute__((address_space(1))) v4i*)g,
      (__attribute__((address_space(3))) v4i*)l, 0, 0);
}
template <int N> static __device__ __forceinline__ void wait_async() {
  __builtin_amdgcn_s_wait_asynccnt(N);
}
#else
static __device__ __forceinline__ void cp16_async(const __bf16* g, __bf16* l) {
  *(uint4*)l = *(const uint4*)g;   // synchronous fallback: reg-staged LDS fill
}
template <int N> static __device__ __forceinline__ void wait_async() {}
#endif

// A fragment: 16(M) x 32(K) bf16, ISA layout: lane = {kh[0],m[3:0]},
// halves 0..7 = K kh*8..kh*8+7, halves 8..15 = K 16+kh*8..+7
static __device__ __forceinline__ v16bf load_afrag(const __bf16* __restrict__ base, int ld, int lane) {
  int m = lane & 15, kh = lane >> 4;
  const __bf16* p = base + (size_t)m * ld + kh * 8;
  v16bf r;
#pragma unroll
  for (int i = 0; i < 8; ++i) { r[i] = p[i]; r[8 + i] = p[16 + i]; }
  return r;
}

// B fragment: 32(K) x 16(N) bf16 with source stored [N, K] row-major:
// lane = {kh[0], n[3:0]}, halves 0..7 = K kh*8..kh*8+7, halves 8..15 = K 16+kh*8..+7 of row n
static __device__ __forceinline__ v16bf load_bfrag(const __bf16* __restrict__ base, int ld, int lane) {
  int n = lane & 15, kh = lane >> 4;
  const __bf16* p = base + (size_t)n * ld + kh * 8;
  v16bf r;
#pragma unroll
  for (int i = 0; i < 8; ++i) { r[i] = p[i]; r[8 + i] = p[16 + i]; }
  return r;
}

static __device__ __forceinline__ v8f wmma_bf16(v16bf a, v16bf b, v8f c) {
  return __builtin_amdgcn_wmma_f32_16x16x32_bf16(false, a, false, b, (short)0, c, false, false);
}

// ---------------------------------------------------------------- converters
__global__ __launch_bounds__(256) void k_cvt(const float* __restrict__ in,
                                             __bf16* __restrict__ out, int n) {
  int i = blockIdx.x * 256 + threadIdx.x;
  if (i < n) out[i] = f2bf(in[i]);
}

// ------------------------------------------------------------ embedding
__global__ __launch_bounds__(256) void k_embed(const int* __restrict__ seq,
                                               const float* __restrict__ emb,
                                               const float* __restrict__ pos,
                                               float* __restrict__ xf,
                                               __bf16* __restrict__ xb) {
  int idx = blockIdx.x * 256 + threadIdx.x;       // over B*S*E
  int e = idx & (EE - 1);
  int bs = idx >> 10;
  int s = bs & (SS - 1);
  int tok = seq[bs];
  float v = emb[(size_t)tok * EE + e] * 32.0f + pos[(size_t)s * EE + e];
  xf[idx] = v;
  xb[idx] = f2bf(v);
}

// ------------------------------------------------------------ generic GEMM
// C[M,N] = A[M,K] * W[N,K]^T + bias. 4 waves/block, 64x64 tile per wave,
// 128x128 block tile. K panels (128x32 A + 128x32 B) double-buffered in LDS,
// filled with async global->LDS copies (ASYNCcnt) overlapped with WMMA.
template <int ACT, bool OF, bool OB>
__global__ __launch_bounds__(128) void k_gemm(
    const __bf16* __restrict__ A, int lda,
    const __bf16* __restrict__ W, int ldw,
    const float* __restrict__ bias,
    float* __restrict__ Cf, __bf16* __restrict__ Cb, int ldc, int K) {
  __shared__ __bf16 sA[2][128 * 32];
  __shared__ __bf16 sB[2][128 * 32];

  int lane = threadIdx.x & 31;
  int w = threadIdx.x >> 5;
  int wm = w >> 1, wn = w & 1;
  int am0 = blockIdx.y * 128;
  int bn0 = blockIdx.x * 128;

  v8f acc[4][4];
#pragma unroll
  for (int mt = 0; mt < 4; ++mt)
#pragma unroll
    for (int nt = 0; nt < 4; ++nt) acc[mt][nt] = vzero8();

  int t = threadIdx.x;
  // panel fill: 128 rows x 32 halves (64B/row) = 512 16B chunks per matrix,
  // 4 chunks per thread per matrix
  auto fill = [&](int buf, int kb) {
#pragma unroll
    for (int i = 0; i < 4; ++i) {
      int flat = i * 128 + t;
      int row = flat >> 2, seg = (flat & 3) * 8;
      cp16_async(A + (size_t)(am0 + row) * lda + kb + seg, &sA[buf][row * 32 + seg]);
    }
#pragma unroll
    for (int i = 0; i < 4; ++i) {
      int flat = i * 128 + t;
      int row = flat >> 2, seg = (flat & 3) * 8;
      cp16_async(W + (size_t)(bn0 + row) * ldw + kb + seg, &sB[buf][row * 32 + seg]);
    }
  };

  fill(0, 0);
  for (int kb = 0; kb < K; kb += 32) {
    int cur = (kb >> 5) & 1;
    bool more = (kb + 32) < K;
    if (more) {
      fill(cur ^ 1, kb + 32);
      wait_async<8>();          // current panel's 8 per-thread copies done
    } else {
      wait_async<0>();
    }
    __syncthreads();

    v16bf af[4], bf[4];
#pragma unroll
    for (int mt = 0; mt < 4; ++mt)
      af[mt] = load_afrag(&sA[cur][(wm * 64 + mt * 16) * 32], 32, lane);
#pragma unroll
    for (int nt = 0; nt < 4; ++nt)
      bf[nt] = load_bfrag(&sB[cur][(wn * 64 + nt * 16) * 32], 32, lane);
#pragma unroll
    for (int mt = 0; mt < 4; ++mt)
#pragma unroll
      for (int nt = 0; nt < 4; ++nt)
        acc[mt][nt] = wmma_bf16(af[mt], bf[nt], acc[mt][nt]);
    __syncthreads();            // protect cur buffer before it is refilled
  }

  int am = am0 + wm * 64, bn = bn0 + wn * 64;
  int lh = lane >> 4, ln = lane & 15;
#pragma unroll
  for (int nt = 0; nt < 4; ++nt) {
    int ncol = bn + nt * 16 + ln;
    float bv = bias ? bias[ncol] : 0.0f;
#pragma unroll
    for (int mt = 0; mt < 4; ++mt) {
#pragma unroll
      for (int r = 0; r < 8; ++r) {
        int row = am + mt * 16 + r + 8 * lh;
        float v = acc[mt][nt][r] + bv;
        if (ACT == 1) v = gelu_exact(v);
        if (OF) Cf[(size_t)row * ldc + ncol] = v;
        if (OB) Cb[(size_t)row * ldc + ncol] = f2bf(v);
      }
    }
  }
}

// ------------------------------------------------ V transpose: vT[b,h,d,s]
__global__ __launch_bounds__(256) void k_vt(const __bf16* __restrict__ qkv,
                                            __bf16* __restrict__ vT) {
  int idx = blockIdx.x * 256 + threadIdx.x;   // over B*H*DH*S
  int s = idx & (SS - 1);
  int d = (idx >> 10) & (DH - 1);
  int h = (idx >> 16) & (HH - 1);
  int b = idx >> 20;
  vT[idx] = qkv[((size_t)(b * SS + s)) * (3 * EE) + 2 * EE + h * DH + d];
}

// ----------------------------------------------------- flash attention
// grid (B*H, S/64); block 128 (4 waves, each wave one 16-row q tile)
__global__ __launch_bounds__(128) void k_attn(const __bf16* __restrict__ qkv,
                                              const __bf16* __restrict__ vT,
                                              const float* __restrict__ dist_emb,
                                              const int* __restrict__ seq,
                                              __bf16* __restrict__ ctx) {
  __shared__ float bias_h[SS];
  __shared__ float kmask[SS];
  __shared__ __bf16 Pt[4][16 * 32];

  int b = blockIdx.x >> 4, h = blockIdx.x & 15;
  for (int i = threadIdx.x; i < SS; i += 128) {
    int d = i < 1024 ? i : 1024;              // min(dist, ML)
    bias_h[i] = dist_emb[(size_t)d * HH + h];
    kmask[i] = (seq[b * SS + i] == PADTOK) ? -1e30f : 0.0f;
  }
  __syncthreads();

  int w = threadIdx.x >> 5, lane = threadIdx.x & 31;
  int lh = lane >> 4, ln = lane & 15;
  int qb = blockIdx.y * 64 + w * 16;

  const __bf16* qrow = qkv + (size_t)(b * SS + qb) * (3 * EE) + h * DH;
  v16bf qf0 = load_afrag(qrow, 3 * EE, lane);
  v16bf qf1 = load_afrag(qrow + 32, 3 * EE, lane);

  v8f o[4];
#pragma unroll
  for (int nt = 0; nt < 4; ++nt) o[nt] = vzero8();
  float mi[8], li[8];
#pragma unroll
  for (int r = 0; r < 8; ++r) { mi[r] = -1e30f; li[r] = 0.0f; }

  const __bf16* kcol = qkv + (size_t)b * SS * (3 * EE) + EE + h * DH;
  const __bf16* vbase = vT + (size_t)(b * HH + h) * DH * SS;

  for (int kb = 0; kb < qb + 16; kb += 32) {
    // prefetch next key block into L2 (global_prefetch_b8)
    if (kb + 32 < qb + 16)
      __builtin_prefetch(kcol + (size_t)(kb + 32) * (3 * EE), 0, 1);

    v8f s0 = vzero8(), s1 = vzero8();
    s0 = wmma_bf16(qf0, load_bfrag(kcol + (size_t)kb * (3 * EE), 3 * EE, lane), s0);
    s0 = wmma_bf16(qf1, load_bfrag(kcol + (size_t)kb * (3 * EE) + 32, 3 * EE, lane), s0);
    s1 = wmma_bf16(qf0, load_bfrag(kcol + (size_t)(kb + 16) * (3 * EE), 3 * EE, lane), s1);
    s1 = wmma_bf16(qf1, load_bfrag(kcol + (size_t)(kb + 16) * (3 * EE) + 32, 3 * EE, lane), s1);

    int k0 = kb + ln, k1 = kb + 16 + ln;
    float msk0 = kmask[k0], msk1 = kmask[k1];
    float p0[8], p1[8], al[8];
#pragma unroll
    for (int r = 0; r < 8; ++r) {
      int q = qb + r + 8 * lh;
      // branchless: clamp the distance index, select -inf via cndmask
      int d0 = q - k0; d0 = d0 < 0 ? 0 : d0;
      int d1 = q - k1; d1 = d1 < 0 ? 0 : d1;
      float b0 = bias_h[d0] + msk0;
      float b1 = bias_h[d1] + msk1;
      float v0 = (k0 <= q) ? s0[r] * 0.125f + b0 : -1e30f;
      float v1 = (k1 <= q) ? s1[r] * 0.125f + b1 : -1e30f;
      float rm = fmaxf(v0, v1);
#pragma unroll
      for (int m = 1; m < 16; m <<= 1) rm = fmaxf(rm, __shfl_xor(rm, m, 32));
      float mnew = fmaxf(mi[r], rm);
      al[r] = __expf(mi[r] - mnew);
      mi[r] = mnew;
      p0[r] = __expf(v0 - mnew);
      p1[r] = __expf(v1 - mnew);
      float rs = p0[r] + p1[r];
#pragma unroll
      for (int m = 1; m < 16; m <<= 1) rs += __shfl_xor(rs, m, 32);
      li[r] = li[r] * al[r] + rs;
    }
#pragma unroll
    for (int nt = 0; nt < 4; ++nt)
#pragma unroll
      for (int r = 0; r < 8; ++r) o[nt][r] *= al[r];

    // stage P (C/D layout) -> LDS -> reload in A-fragment layout
    __bf16* P = Pt[w];
#pragma unroll
    for (int r = 0; r < 8; ++r) {
      P[(r + 8 * lh) * 32 + ln] = f2bf(p0[r]);
      P[(r + 8 * lh) * 32 + 16 + ln] = f2bf(p1[r]);
    }
    asm volatile("s_wait_dscnt 0" ::: "memory");   // in-wave LDS RAW ordering
    v16bf pf;
    {
      const __bf16* pp = Pt[w] + (lane & 15) * 32 + (lane >> 4) * 8;
#pragma unroll
      for (int i = 0; i < 8; ++i) { pf[i] = pp[i]; pf[8 + i] = pp[16 + i]; }
    }
#pragma unroll
    for (int nt = 0; nt < 4; ++nt)
      o[nt] = wmma_bf16(pf, load_bfrag(vbase + (size_t)(nt * 16) * SS + kb, SS, lane), o[nt]);
  }

#pragma unroll
  for (int r = 0; r < 8; ++r) {
    float inv = 1.0f / li[r];
    int q = qb + r + 8 * lh;
#pragma unroll
    for (int nt = 0; nt < 4; ++nt)
      ctx[(size_t)(b * SS + q) * EE + h * DH + nt * 16 + ln] = f2bf(o[nt][r] * inv);
  }
}

// ---------------------------------------------- residual add + LayerNorm
__global__ __launch_bounds__(256) void k_addln(const float* __restrict__ xin,
                                               const float* __restrict__ res,
                                               const float* __restrict__ g,
                                               const float* __restrict__ bta,
                                               float* __restrict__ xf,
                                               __bf16* __restrict__ xb) {
  __shared__ float red[256];
  int row = blockIdx.x;
  int tid = threadIdx.x;
  const float* xr = xin + (size_t)row * EE;
  const float* rr = res ? res + (size_t)row * EE : nullptr;
  float v[4];
  float s = 0.0f;
#pragma unroll
  for (int i = 0; i < 4; ++i) {
    int e = tid + i * 256;
    v[i] = xr[e] + (rr ? rr[e] : 0.0f);
    s += v[i];
  }
  red[tid] = s; __syncthreads();
  for (int off = 128; off > 0; off >>= 1) {
    if (tid < off) red[tid] += red[tid + off];
    __syncthreads();
  }
  float mean = red[0] * (1.0f / EE);
  __syncthreads();
  s = 0.0f;
#pragma unroll
  for (int i = 0; i < 4; ++i) { float d = v[i] - mean; s += d * d; }
  red[tid] = s; __syncthreads();
  for (int off = 128; off > 0; off >>= 1) {
    if (tid < off) red[tid] += red[tid + off];
    __syncthreads();
  }
  float rinv = rsqrtf(red[0] * (1.0f / EE) + 1e-5f);
#pragma unroll
  for (int i = 0; i < 4; ++i) {
    int e = tid + i * 256;
    float ov = (v[i] - mean) * rinv * g[e] + bta[e];
    xf[(size_t)row * EE + e] = ov;
    xb[(size_t)row * EE + e] = f2bf(ov);
  }
}

// -------------------------------------------------------------- launcher
extern "C" void kernel_launch(void* const* d_in, const int* in_sizes, int n_in,
                              void* d_out, int out_size, void* d_ws, size_t ws_size,
                              hipStream_t stream) {
  (void)in_sizes; (void)n_in; (void)out_size; (void)ws_size;
  const int*   seq    = (const int*)d_in[0];
  const float* emb    = (const float*)d_in[1];
  const float* pos    = (const float*)d_in[2];
  const float* dist   = (const float*)d_in[3];
  const float* in_w   = (const float*)d_in[4];
  const float* in_b   = (const float*)d_in[5];
  const float* out_w  = (const float*)d_in[6];
  const float* out_b  = (const float*)d_in[7];
  const float* ln1g   = (const float*)d_in[8];
  const float* ln1b   = (const float*)d_in[9];
  const float* ff1_w  = (const float*)d_in[10];
  const float* ff1_b  = (const float*)d_in[11];
  const float* ff2_w  = (const float*)d_in[12];
  const float* ff2_b  = (const float*)d_in[13];
  const float* ln2g   = (const float*)d_in[14];
  const float* ln2b   = (const float*)d_in[15];
  const float* lnfg   = (const float*)d_in[16];
  const float* lnfb   = (const float*)d_in[17];
  const float* gen_w  = (const float*)d_in[18];
  const float* gen_b  = (const float*)d_in[19];
  float* out = (float*)d_out;

  char* ws = (char*)d_ws;
  size_t off = 0;
  auto alloc = [&](size_t bytes) -> void* {
    void* p = ws + off;
    off += (bytes + 255) & ~(size_t)255;
    return p;
  };
  __bf16* wb_in  = (__bf16*)alloc((size_t)LL * 3 * EE * EE * 2);
  __bf16* wb_out = (__bf16*)alloc((size_t)LL * EE * EE * 2);
  __bf16* wb_ff1 = (__bf16*)alloc((size_t)LL * FF * EE * 2);
  __bf16* wb_ff2 = (__bf16*)alloc((size_t)LL * EE * FF * 2);
  __bf16* wb_gen = (__bf16*)alloc((size_t)EE * EE * 2);
  float*  xf     = (float*)alloc((size_t)BB * SS * EE * 4);
  __bf16* xb     = (__bf16*)alloc((size_t)BB * SS * EE * 2);
  __bf16* qkvb   = (__bf16*)alloc((size_t)BB * SS * 3 * EE * 2);
  __bf16* vtb    = (__bf16*)alloc((size_t)BB * HH * DH * SS * 2);
  __bf16* ctxb   = (__bf16*)alloc((size_t)BB * SS * EE * 2);
  float*  tmpf   = (float*)alloc((size_t)BB * SS * EE * 4);
  __bf16* hffb   = (__bf16*)alloc((size_t)BB * SS * FF * 2);

  const int M = BB * SS;  // 4096

  // weight conversion to bf16 (deterministic each launch)
  k_cvt<<<(LL * 3 * EE * EE) / 256, 256, 0, stream>>>(in_w, wb_in, LL * 3 * EE * EE);
  k_cvt<<<(LL * EE * EE) / 256, 256, 0, stream>>>(out_w, wb_out, LL * EE * EE);
  k_cvt<<<(LL * FF * EE) / 256, 256, 0, stream>>>(ff1_w, wb_ff1, LL * FF * EE);
  k_cvt<<<(LL * EE * FF) / 256, 256, 0, stream>>>(ff2_w, wb_ff2, LL * EE * FF);
  k_cvt<<<(EE * EE) / 256, 256, 0, stream>>>(gen_w, wb_gen, EE * EE);

  k_embed<<<(M * EE) / 256, 256, 0, stream>>>(seq, emb, pos, xf, xb);

  for (int l = 0; l < LL; ++l) {
    // QKV projection -> bf16
    k_gemm<0, false, true><<<dim3(3 * EE / 128, M / 128), 128, 0, stream>>>(
        xb, EE, wb_in + (size_t)l * 3 * EE * EE, EE, in_b + l * 3 * EE,
        nullptr, qkvb, 3 * EE, EE);
    // V transpose for WMMA B-fragments
    k_vt<<<(BB * HH * DH * SS) / 256, 256, 0, stream>>>(qkvb, vtb);
    // flash attention
    k_attn<<<dim3(BB * HH, SS / 64), 128, 0, stream>>>(qkvb, vtb, dist, seq, ctxb);
    // output projection -> f32
    k_gemm<0, true, false><<<dim3(EE / 128, M / 128), 128, 0, stream>>>(
        ctxb, EE, wb_out + (size_t)l * EE * EE, EE, out_b + l * EE,
        tmpf, nullptr, EE, EE);
    k_addln<<<M, 256, 0, stream>>>(xf, tmpf, ln1g + l * EE, ln1b + l * EE, xf, xb);
    // FF1 + GELU -> bf16
    k_gemm<1, false, true><<<dim3(FF / 128, M / 128), 128, 0, stream>>>(
        xb, EE, wb_ff1 + (size_t)l * FF * EE, EE, ff1_b + l * FF,
        nullptr, hffb, FF, EE);
    // FF2 -> f32
    k_gemm<0, true, false><<<dim3(EE / 128, M / 128), 128, 0, stream>>>(
        hffb, FF, wb_ff2 + (size_t)l * EE * FF, FF, ff2_b + l * EE,
        tmpf, nullptr, EE, FF);
    k_addln<<<M, 256, 0, stream>>>(xf, tmpf, ln2g + l * EE, ln2b + l * EE, xf, xb);
  }
  // final LN (no residual)
  k_addln<<<M, 256, 0, stream>>>(xf, nullptr, lnfg, lnfb, xf, xb);
  // logits
  k_gemm<0, true, false><<<dim3(EE / 128, M / 128), 128, 0, stream>>>(
      xb, EE, wb_gen, EE, gen_b, out, nullptr, EE, EE);
}
